// SinenetLayerV3_34437047779484
// MI455X (gfx1250) — compile-verified, matrix operand order checked by reading, not run on hardware
//
#include <hip/hip_runtime.h>
#include <hip/hip_bf16.h>
#include <math.h>

typedef __attribute__((ext_vector_type(16))) _Float16 v16h;
typedef __attribute__((ext_vector_type(8)))  float    v8f;

#define S_DIM 16
#define B_DIM 16
#define M_DIM 20
#define T_DIM 640
#define K_DIM 16
#define D_DIM 80
#define OUTW  (M_DIM * D_DIM + M_DIM)   // 1620
#define NROW  (S_DIM * B_DIM * M_DIM)   // 5120

__device__ __forceinline__ float relu(float v) { return v > 0.f ? v : 0.f; }

__global__ __launch_bounds__(512) void sinenet_kernel(
    const float* __restrict__ x, const float* __restrict__ nlf,
    const float* __restrict__ tau, const float* __restrict__ W,
    const float* __restrict__ bvec, float* __restrict__ out)
{
    constexpr float T_WAV      = 1.0f / 16000.0f;
    constexpr float LOG_F_MEAN = 5.04418f;
    constexpr float LOG_F_STD  = 0.358402f;
    constexpr float TWO_PI     = 6.283185307179586f;

    // A-tile for WMMA: sin_cos_x for 16 rows, 2K=32 columns, f16.
    __shared__ _Float16 Atile[16][32];

    const int lane = threadIdx.x & 31;
    const int wv   = threadIdx.x >> 5;          // 16 waves
    const int tileBase = blockIdx.x * 16;       // 16 (s,b,m) rows per block

    // ---------------- Phase 1: wave wv -> row tileBase+wv ----------------
    {
        const int row = tileBase + wv;
        const float f     = __expf(nlf[row] * LOG_F_STD + LOG_F_MEAN);
        const float omega = TWO_PI * f;
        const float tv    = tau[row];
        const float* xr   = x + (size_t)row * T_DIM;

        float accS[K_DIM], accC[K_DIM];
#pragma unroll
        for (int k = 0; k < K_DIM; ++k) { accS[k] = 0.f; accC[k] = 0.f; }

        // lanes partition T with stride 32 -> coalesced 128B x loads.
        for (int n = lane; n < T_DIM; n += 32) {
            const float xv = xr[n];
            const float th = omega * (n * T_WAV - tv);
            float s1, c1;
            __sincosf(th, &s1, &c1);            // one sincos per (row,t)
            float sk = s1, ck = c1;
            accS[0] += s1 * xv;
            accC[0] += c1 * xv;
#pragma unroll
            for (int k = 1; k < K_DIM; ++k) {   // angle-addition recurrence
                const float sn = __fmaf_rn(sk, c1,  ck * s1);
                const float cn = __fmaf_rn(ck, c1, -sk * s1);
                sk = sn; ck = cn;
                accS[k] += sk * xv;             // sin & cos chains independent
                accC[k] += ck * xv;             // -> VOPD dual-issue fodder
            }
        }

        // ---- transpose-reduction: 32 values x 32 lanes, 31 shuffles ----
        // Invariant: final-index bit b == lane bit b. Lane l ends with
        // total[l]; l<16 -> s_x[l], l>=16 -> c_x[l-16] (matches W layout).
        float r16[16];
        {
            const bool hb = (lane & 16) != 0;
#pragma unroll
            for (int j = 0; j < 16; ++j) {
                const float send = hb ? accS[j] : accC[j];
                const float keep = hb ? accC[j] : accS[j];
                r16[j] = keep + __shfl_xor(send, 16, 32);
            }
        }
        float r8[8];
        {
            const bool hb = (lane & 8) != 0;
#pragma unroll
            for (int j = 0; j < 8; ++j) {
                const float send = hb ? r16[j] : r16[j + 8];
                const float keep = hb ? r16[j + 8] : r16[j];
                r8[j] = keep + __shfl_xor(send, 8, 32);
            }
        }
        float r4[4];
        {
            const bool hb = (lane & 4) != 0;
#pragma unroll
            for (int j = 0; j < 4; ++j) {
                const float send = hb ? r8[j] : r8[j + 4];
                const float keep = hb ? r8[j + 4] : r8[j];
                r4[j] = keep + __shfl_xor(send, 4, 32);
            }
        }
        float r2[2];
        {
            const bool hb = (lane & 2) != 0;
#pragma unroll
            for (int j = 0; j < 2; ++j) {
                const float send = hb ? r4[j] : r4[j + 2];
                const float keep = hb ? r4[j + 2] : r4[j];
                r2[j] = keep + __shfl_xor(send, 2, 32);
            }
        }
        float total;
        {
            const bool hb = (lane & 1) != 0;
            const float send = hb ? r2[0] : r2[1];
            const float keep = hb ? r2[1] : r2[0];
            total = keep + __shfl_xor(send, 1, 32);
        }
        // all 32 lanes store their element in parallel
        Atile[wv][lane] = (_Float16)total;
    }
    __syncthreads();

    // ---------------- Phase 2: waves 0..4 -> WMMA fc tiles ----------------
    if (wv < 5) {
        // wave-uniform branch: EXEC is all-ones inside (WMMA requirement)
        const int hi = lane >> 4;       // which K half this lane holds
        const int n  = lane & 15;       // A: M index; B: N index
        const int d  = wv * 16 + n;     // output feature column

        v16h a, bf;
#pragma unroll
        for (int e = 0; e < 16; ++e) {
            // 16-bit fragment K packing (ISA 7.12.2):
            // elems 0..7 -> K = hi*8 + e ; elems 8..15 -> K = 16 + hi*8 + (e-8)
            const int k = (e < 8) ? (hi * 8 + e) : (16 + hi * 8 + (e - 8));
            a[e]  = Atile[n][k];                // contiguous -> ds_load_b128
            bf[e] = (_Float16)W[d * 32 + k];    // contiguous -> global b128
        }
        v8f cz = {};
        v8f c = __builtin_amdgcn_wmma_f32_16x16x32_f16(
            /*neg_a=*/false, a, /*neg_b=*/false, bf,
            /*c_mod=*/(short)0, cz, /*reuse_a=*/false, /*reuse_b=*/false);

        const float bval = bvec[d];
#pragma unroll
        for (int r = 0; r < 8; ++r) {
            const int m  = hi * 8 + r;             // C/D: M = vgpr idx (+8 hi)
            const int g  = tileBase + m;           // global sbm row
            const int sb = g / M_DIM;
            const int mi = g - sb * M_DIM;
            out[(size_t)sb * OUTW + mi * D_DIM + d] = relu(c[r] + bval);
        }
    } else if (wv == 5 && lane < 16) {
        // appended relu(nlf) tail columns
        const int g  = tileBase + lane;
        const int sb = g / M_DIM;
        const int mi = g - sb * M_DIM;
        out[(size_t)sb * OUTW + M_DIM * D_DIM + mi] = relu(nlf[g]);
    }
}

extern "C" void kernel_launch(void* const* d_in, const int* in_sizes, int n_in,
                              void* d_out, int out_size, void* d_ws, size_t ws_size,
                              hipStream_t stream) {
    const float* x   = (const float*)d_in[0];
    const float* nlf = (const float*)d_in[1];
    const float* tau = (const float*)d_in[2];
    const float* W   = (const float*)d_in[3];
    const float* b   = (const float*)d_in[4];
    float* out = (float*)d_out;

    dim3 grid(NROW / 16);   // 320 workgroups
    dim3 block(512);        // 16 wave32 waves
    hipLaunchKernelGGL(sinenet_kernel, grid, block, 0, stream,
                       x, nlf, tau, W, b, out);
}
